// DifferentiablePooler_43868795961667
// MI455X (gfx1250) — matrix operand order repaired and checked by reading, not compiled, not run on hardware
//
#include <hip/hip_runtime.h>
#include <stdint.h>

// MI455X / gfx1250, wave32. Split-bf16 WMMA GEMM pipeline for
// relu(C @ relu(Dinv2 A Dinv2 @ (X W_in) + b_in) @ W_out + b_out) pooler.

typedef __attribute__((ext_vector_type(16))) __bf16 v16bf;
typedef __attribute__((ext_vector_type(8)))  float  v8f;

#define NN  8192   // nodes
#define NCC 4096   // coarse nodes
#define NF  128    // features

static __device__ __forceinline__ unsigned short f32_to_bf16_rne(float f) {
  unsigned u = __float_as_uint(f);
  unsigned r = 0x7FFFu + ((u >> 16) & 1u);
  return (unsigned short)((u + r) >> 16);
}
static __device__ __forceinline__ float bf16_bits_to_f32(unsigned short h) {
  return __uint_as_float(((unsigned)h) << 16);
}
static __device__ __forceinline__ unsigned pack_hi2(float a, float b) {
  return (unsigned)f32_to_bf16_rne(a) | ((unsigned)f32_to_bf16_rne(b) << 16);
}

// Packed B-operand layout for v_wmma_f32_16x16x32_bf16:
// block (kb = k/32, nt = f/16) of 32x16 stored as 32 lanes x 16 bf16 (32B/lane).
// lane = (f%16) + 16*((k%32)/16), element-in-lane = k%16.
static __device__ __forceinline__ size_t bpack_idx(int k, int f) {
  const int kb = k >> 5, kp = k & 31;
  const int nt = f >> 4, nn = f & 15;
  return (((size_t)(kb * 8 + nt)) << 9) +
         (size_t)(((nn + ((kp >> 4) << 4)) << 4) + (kp & 15));
}

// ---------------- Kernel 1: deg = rowsum(A); ds = rsqrt ----------------
__global__ __launch_bounds__(256) void deg_kernel(const float* __restrict__ A,
                                                  float* __restrict__ ds) {
  const int row = blockIdx.x;
  const float4* __restrict__ Ar = (const float4*)(A + (size_t)row * NN);
  float s = 0.f;
  for (int c = threadIdx.x; c < NN / 4; c += 256) {
    float4 v = Ar[c];
    s += (v.x + v.y) + (v.z + v.w);
  }
  __shared__ float red[256];
  red[threadIdx.x] = s;
  __syncthreads();
  for (int o = 128; o > 0; o >>= 1) {
    if (threadIdx.x < o) red[threadIdx.x] += red[threadIdx.x + o];
    __syncthreads();
  }
  if (threadIdx.x == 0) {
    float d = red[0];
    ds[row] = d > 0.f ? rsqrtf(fmaxf(d, 1e-12f)) : 0.f;
  }
}

// ------- Kernel 2: XW'[j,f] = ds[j]*(X@W_in)[j,f], split hi/lo, B-packed -------
__global__ __launch_bounds__(128) void xw_pack_kernel(
    const float* __restrict__ X, const float* __restrict__ Win,
    const float* __restrict__ ds,
    unsigned short* __restrict__ Bhi, unsigned short* __restrict__ Blo) {
  const int j = blockIdx.x;
  const int f = threadIdx.x;
  __shared__ float xr[NF];
  xr[f] = X[(size_t)j * NF + f];
  __syncthreads();
  float acc = 0.f;
#pragma unroll 16
  for (int k = 0; k < NF; ++k) acc = fmaf(xr[k], Win[k * NF + f], acc);
  const float v = ds[j] * acc;
  const unsigned short hb = f32_to_bf16_rne(v);
  const unsigned short lb = f32_to_bf16_rne(v - bf16_bits_to_f32(hb));
  const size_t idx = bpack_idx(j, f);
  Bhi[idx] = hb;
  Blo[idx] = lb;
}

// ------- Kernels 3/4: split-bf16 WMMA GEMM, 64x128 tile per 256-thread block -------
// D = A(fp32, streamed, split to hi/lo bf16 on the fly) @ Bpacked(hi/lo bf16)
// 3 x v_wmma_f32_16x16x32_bf16 per k-step (hi*hi + lo*hi + hi*lo), fp32 accum.
// Double-buffered LDS staging (1 barrier/k-step), B loads hoisted ahead of the
// stage so L2 latency overlaps conversion work.
template <bool SCALE_ROWS, bool EPI_PACK>
__global__ __launch_bounds__(256) void gemm_split_bf16(
    const float* __restrict__ Amat,            // [M x 8192] row-major fp32
    const unsigned short* __restrict__ Bhi,    // packed B hi
    const unsigned short* __restrict__ Blo,    // packed B lo
    const float* __restrict__ ds,              // row scale (SCALE_ROWS)
    const float* __restrict__ bias,            // EPI_PACK: + bias, relu
    float* __restrict__ Cout,                  // !EPI_PACK: [M x 128] fp32
    unsigned short* __restrict__ OutHi,        // EPI_PACK: packed bf16 hi
    unsigned short* __restrict__ OutLo) {      // EPI_PACK: packed bf16 lo
  constexpr int K = NN;
  constexpr int KB = K / 32;
  const int tid  = threadIdx.x;
  const int lane = tid & 31;
  const int wv   = tid >> 5;        // 0..7
  const int rt   = wv & 3;          // row-tile of 16 rows
  const int ntb  = (wv >> 2) << 2;  // n-tile base: 0 or 4
  const size_t row0 = (size_t)blockIdx.x * 64;

  // double-buffered A staging: 2 x (4 tiles of 16x32 bf16) for hi and lo
  __shared__ alignas(32) unsigned short sAhi[2 * 2048];
  __shared__ alignas(32) unsigned short sAlo[2 * 2048];

  // Stage A 64x32 fp32 (tile kbs) -> hi/lo bf16 in WMMA A-operand layout.
  auto stage = [&](int kbs, int buf) {
    unsigned short* dHi = sAhi + (buf << 11);
    unsigned short* dLo = sAlo + (buf << 11);
#pragma unroll
    for (int i = 0; i < 2; ++i) {
      const int q  = tid + (i << 8);   // 0..511 quad index
      const int m  = q >> 3;           // 0..63 row in tile
      const int kq = (q & 7) << 2;     // k: 0,4,...,28
      const float4 v = *(const float4*)(Amat + (row0 + m) * (size_t)K +
                                        (size_t)(kbs << 5) + kq);
      const unsigned h01 = pack_hi2(v.x, v.y);
      const unsigned h23 = pack_hi2(v.z, v.w);
      const unsigned l01 =
          pack_hi2(v.x - bf16_bits_to_f32((unsigned short)h01),
                   v.y - bf16_bits_to_f32((unsigned short)(h01 >> 16)));
      const unsigned l23 =
          pack_hi2(v.z - bf16_bits_to_f32((unsigned short)h23),
                   v.w - bf16_bits_to_f32((unsigned short)(h23 >> 16)));
      // A-layout (contiguous k, matches B pack): lane = m%16 + 16*(k>>4); slot = k&15
      const int ln   = (m & 15) + ((kq >> 4) << 4);
      const int slot = kq & 15;   // {0,4,8,12}
      const int base = ((m >> 4) << 9) + (ln << 4) + slot;  // ushort units
      *(uint2*)&dHi[base] = uint2{h01, h23};
      *(uint2*)&dLo[base] = uint2{l01, l23};
    }
  };

  v8f acc[4];
#pragma unroll
  for (int i = 0; i < 4; ++i) acc[i] = v8f{0.f,0.f,0.f,0.f,0.f,0.f,0.f,0.f};

  stage(0, 0);
  __syncthreads();

  for (int kb = 0; kb < KB; ++kb) {
    const int buf = kb & 1;
    // --- hoist B loads (L2-resident packed bf16): latency overlaps staging ---
    const unsigned short* __restrict__ bph = Bhi + ((size_t)kb << 12) + (lane << 4);
    const unsigned short* __restrict__ bpl = Blo + ((size_t)kb << 12) + (lane << 4);
    v16bf bh[4], bl[4];
#pragma unroll
    for (int j = 0; j < 4; ++j) {
      bh[j] = *(const v16bf*)(bph + ((size_t)(ntb + j) << 9));
      bl[j] = *(const v16bf*)(bpl + ((size_t)(ntb + j) << 9));
    }
    // --- A operands from current LDS buffer ---
    const int aoff = (buf << 11) + (rt << 9) + (lane << 4);
    const v16bf ah = *(const v16bf*)&sAhi[aoff];
    const v16bf al = *(const v16bf*)&sAlo[aoff];
    // --- stage next tile into the other buffer while WMMAs run ---
    if (kb + 1 < KB) stage(kb + 1, buf ^ 1);
    // --- prefetch tile kb+2 (global_prefetch_b8), 256 threads x 32B = 8KB ---
    if (kb + 2 < KB) {
      const float* pf = Amat + (row0 + (tid >> 2)) * (size_t)K +
                        (size_t)((kb + 2) << 5) + ((tid & 3) << 3);
      __builtin_prefetch(pf, 0, 1);
    }
    // --- 12 WMMAs: hi*hi + lo*hi + hi*lo ---
#pragma unroll
    for (int j = 0; j < 4; ++j)
      acc[j] = __builtin_amdgcn_wmma_f32_16x16x32_bf16(
          false, ah, false, bh[j], (short)0, acc[j], false, false);
#pragma unroll
    for (int j = 0; j < 4; ++j)
      acc[j] = __builtin_amdgcn_wmma_f32_16x16x32_bf16(
          false, al, false, bh[j], (short)0, acc[j], false, false);
#pragma unroll
    for (int j = 0; j < 4; ++j)
      acc[j] = __builtin_amdgcn_wmma_f32_16x16x32_bf16(
          false, ah, false, bl[j], (short)0, acc[j], false, false);
    __syncthreads();
  }

  // Epilogue. D layout: VGPR g -> row g + 8*(lane/16); col = lane%16.
  const int nloc = lane & 15;
  const int mh   = (lane >> 4) << 3;          // 0 or 8
  const size_t ibase = row0 + (size_t)((rt << 4) + mh);  // 8-aligned row base
  if constexpr (EPI_PACK) {
    float dsv[8];
#pragma unroll
    for (int g = 0; g < 8; ++g) dsv[g] = SCALE_ROWS ? ds[ibase + g] : 1.f;
#pragma unroll
    for (int j = 0; j < 4; ++j) {
      const int f = ((ntb + j) << 4) + nloc;
      const float bv = bias[f];
      unsigned uh[4], ul[4];
#pragma unroll
      for (int g = 0; g < 8; g += 2) {
        float v0 = fmaxf(acc[j][g] * dsv[g] + bv, 0.f);
        float v1 = fmaxf(acc[j][g + 1] * dsv[g + 1] + bv, 0.f);
        const unsigned h = pack_hi2(v0, v1);
        uh[g >> 1] = h;
        ul[g >> 1] = pack_hi2(v0 - bf16_bits_to_f32((unsigned short)h),
                              v1 - bf16_bits_to_f32((unsigned short)(h >> 16)));
      }
      // 8 consecutive rows of one lane land on 8 consecutive packed elements
      const size_t o0 = bpack_idx((int)ibase, f);   // 16B-aligned
      *(uint4*)(OutHi + o0) = uint4{uh[0], uh[1], uh[2], uh[3]};
      *(uint4*)(OutLo + o0) = uint4{ul[0], ul[1], ul[2], ul[3]};
    }
  } else {
#pragma unroll
    for (int j = 0; j < 4; ++j) {
      const int f = ((ntb + j) << 4) + nloc;
#pragma unroll
      for (int g = 0; g < 8; ++g) {
        float v = acc[j][g];
        if constexpr (SCALE_ROWS) v *= ds[ibase + g];
        Cout[(ibase + g) * NF + f] = v;
      }
    }
  }
}

// ------- Kernel 5: out = relu(Xp @ W_out + b_out) (tiny, VALU) -------
__global__ __launch_bounds__(128) void dense_out_kernel(
    const float* __restrict__ Xp, const float* __restrict__ W,
    const float* __restrict__ b, float* __restrict__ out) {
  const int i = blockIdx.x;
  const int f = threadIdx.x;
  __shared__ float xr[NF];
  xr[f] = Xp[(size_t)i * NF + f];
  __syncthreads();
  float acc = b[f];
#pragma unroll 16
  for (int k = 0; k < NF; ++k) acc = fmaf(xr[k], W[k * NF + f], acc);
  out[(size_t)i * NF + f] = fmaxf(acc, 0.f);
}

extern "C" void kernel_launch(void* const* d_in, const int* in_sizes, int n_in,
                              void* d_out, int out_size, void* d_ws, size_t ws_size,
                              hipStream_t stream) {
  (void)in_sizes; (void)n_in; (void)out_size; (void)ws_size;
  const float* X    = (const float*)d_in[0];   // [8192,128]
  const float* A    = (const float*)d_in[1];   // [8192,8192]
  const float* Cm   = (const float*)d_in[2];   // [4096,8192]
  const float* Win  = (const float*)d_in[3];   // [128,128]
  const float* bin  = (const float*)d_in[4];   // [128]
  const float* Wout = (const float*)d_in[5];   // [128,128]
  const float* bout = (const float*)d_in[6];   // [128]

  char* ws = (char*)d_ws;
  float* ds            = (float*)ws;                                // 32 KB
  unsigned short* XWhi = (unsigned short*)(ws + (32 << 10));        // 2 MB
  unsigned short* XWlo = XWhi + (size_t)NN * NF;                    // 2 MB
  unsigned short* Hhi  = XWlo + (size_t)NN * NF;                    // 2 MB
  unsigned short* Hlo  = Hhi  + (size_t)NN * NF;                    // 2 MB
  float* Xp            = (float*)(Hlo + (size_t)NN * NF);           // 2 MB

  // 1) ds = rsqrt(rowsum(A))
  deg_kernel<<<NN, 256, 0, stream>>>(A, ds);
  // 2) XW' = ds[j] * (X @ W_in), split-bf16, packed B layout (L2-resident)
  xw_pack_kernel<<<NN, NF, 0, stream>>>(X, Win, ds, XWhi, XWlo);
  // 3) H = relu(ds[i]*(A @ XW') + b_in) -> packed bf16 hi/lo (B of next GEMM)
  gemm_split_bf16<true, true><<<NN / 64, 256, 0, stream>>>(
      A, XWhi, XWlo, ds, bin, nullptr, Hhi, Hlo);
  // 4) Xp = C @ H -> fp32
  gemm_split_bf16<false, false><<<NCC / 64, 256, 0, stream>>>(
      Cm, Hhi, Hlo, nullptr, nullptr, Xp, nullptr, nullptr);
  // 5) out = relu(Xp @ W_out + b_out)
  dense_out_kernel<<<NCC, NF, 0, stream>>>(Xp, Wout, bout, (float*)d_out);
}
